// MultiHeadAttention_50929722196246
// MI455X (gfx1250) — matrix operand order repaired and checked by reading, not compiled, not run on hardware
//
#include <hip/hip_runtime.h>
#include <hip/hip_bf16.h>
#include <math.h>

// ---------------------------------------------------------------------------
// MHA decode for MI455X (gfx1250, wave32).
// Memory-bound (AI ~8 FLOP/B): stream fp32 KV cache once, bf16 WMMA math,
// fp32 accumulation. All matrix math via v_wmma_f32_16x16x32_bf16.
// ---------------------------------------------------------------------------

typedef __bf16 v16bf __attribute__((ext_vector_type(16)));
typedef float  v8f   __attribute__((ext_vector_type(8)));

#define NH_    16
#define HD_    128
#define BB_    16
#define SS_    8
#define HH_    2048
#define KV_    4096
#define KVTOT  (KV_ + SS_)     // 4104
#define ROWS_  (BB_ * SS_)     // 128

// ---------------------------------------------------------------------------
// One 16x16 output tile of C = bf16(X[128 x 2048]) * bf16(W[2048 x 2048]),
// fp32 accum.  Wave-level: A fragment (16x32 bf16) row = lane&15, K split by
// lane half; B fragment (32x16 bf16) col = lane&15, K half = lane>>4.
// ---------------------------------------------------------------------------
__device__ __forceinline__ v8f gemm_tile_bf16(const float* __restrict__ X,
                                              const float* __restrict__ W,
                                              int m0, int n0)
{
    const int lane = threadIdx.x & 31;
    const int hf   = lane >> 4;
    const int l15  = lane & 15;

    v8f acc;
#pragma unroll
    for (int r = 0; r < 8; ++r) acc[r] = 0.0f;

    const float* xrow = X + (m0 + l15) * HH_;
    const float* wcol = W + n0 + l15;

    for (int k = 0; k < HH_; k += 32) {
        // A: lane half 0 -> K = k+0..15 ; half 1 -> K = k+16..31
        const float4 s0 = *(const float4*)(xrow + k + hf * 16);
        const float4 s1 = *(const float4*)(xrow + k + hf * 16 + 4);
        const float4 s2 = *(const float4*)(xrow + k + hf * 16 + 8);
        const float4 s3 = *(const float4*)(xrow + k + hf * 16 + 12);
        v16bf a;
        a[0]=(__bf16)s0.x;  a[1]=(__bf16)s0.y;  a[2]=(__bf16)s0.z;  a[3]=(__bf16)s0.w;
        a[4]=(__bf16)s1.x;  a[5]=(__bf16)s1.y;  a[6]=(__bf16)s1.z;  a[7]=(__bf16)s1.w;
        a[8]=(__bf16)s2.x;  a[9]=(__bf16)s2.y;  a[10]=(__bf16)s2.z; a[11]=(__bf16)s2.w;
        a[12]=(__bf16)s3.x; a[13]=(__bf16)s3.y; a[14]=(__bf16)s3.z; a[15]=(__bf16)s3.w;

        // B: element j -> K = k + 16*half + j (column n0+l15, row stride 2048)
        const float* wc = wcol + (size_t)(k + hf * 16) * HH_;
        __builtin_prefetch(wc + (size_t)32 * HH_, 0, 1);
        v16bf wb;
#pragma unroll
        for (int j = 0; j < 16; ++j) wb[j] = (__bf16)wc[(size_t)j * HH_];

        acc = __builtin_amdgcn_wmma_f32_16x16x32_bf16(false, a, false, wb,
                                                      (short)0, acc, false, false);
    }
    return acc;
}

// ---------------------------------------------------------------------------
// Kernel 1: fused QKV projection.  Output written transposed to [B,nh,S,hd].
// 3072 tiles (8 row-tiles x 384 col-tiles over Wq|Wk|Wv), 4 waves / block.
// ---------------------------------------------------------------------------
__global__ void __launch_bounds__(128) qkv_gemm_kernel(
    const float* __restrict__ x,
    const float* __restrict__ Wq, const float* __restrict__ Wk,
    const float* __restrict__ Wv,
    float* __restrict__ qb, float* __restrict__ kb, float* __restrict__ vb)
{
    const int wave = threadIdx.x >> 5;
    const int tile = blockIdx.x * 4 + wave;   // 0..3071
    const int nt = tile >> 3;                 // 0..383
    const int mt = tile & 7;
    const int m0 = mt * 16;
    const int n0g = nt * 16;                  // global col in [0,6144)
    const int which = n0g >> 11;              // 0=Q 1=K 2=V
    const int n0 = n0g & 2047;

    const float* W = (which == 0) ? Wq : (which == 1) ? Wk : Wv;
    v8f acc = gemm_tile_bf16(x, W, m0, n0);

    const int lane = threadIdx.x & 31;
    const int hf = lane >> 4, l15 = lane & 15;
    const int c  = n0 + l15;
    const int hh = c >> 7;                    // head
    const int dd = c & 127;                   // dim within head
    float* dst = (which == 0) ? qb : (which == 1) ? kb : vb;

#pragma unroll
    for (int r = 0; r < 8; ++r) {
        const int m = m0 + r + 8 * hf;        // D layout: M = r + 8*(lane>>4)
        const int bi = m >> 3, si = m & 7;
        dst[(((bi * NH_) + hh) * SS_ + si) * HD_ + dd] = acc[r];
    }
}

// ---------------------------------------------------------------------------
// Kernel 2: RoPE in-place on q and k_new (positions 0..S-1).
// idx < 131072 -> q, else k.  Each thread handles one (b,h,s,d<64) pair.
// ---------------------------------------------------------------------------
__global__ void __launch_bounds__(256) rope_kernel(float* __restrict__ qb,
                                                   float* __restrict__ kb)
{
    const int idx = blockIdx.x * blockDim.x + threadIdx.x;   // < 262144
    float* arr = (idx < 131072) ? qb : kb;
    const int i  = idx & (131072 - 1);
    const int d  = i & 63;
    const int s  = (i >> 6) & 7;
    const int bh = i >> 9;

    // inv_freq = 10000^(-2d/128) = exp(-2d/128 * ln(10000))
    const float inv = __expf(-(float)(2 * d) * (9.210340371976184f / 128.0f));
    const float f = (float)s * inv;
    float sn, cs;
    __sincosf(f, &sn, &cs);

    float* row = arr + (bh * SS_ + s) * HD_;
    const float x1 = row[d], x2 = row[d + 64];
    row[d]      = x1 * cs - x2 * sn;
    row[d + 64] = x2 * cs + x1 * sn;
}

__device__ __forceinline__ float load_v_elem(const float* __restrict__ Vp,
                                             const float* __restrict__ Vn,
                                             int kv, int d)
{
    if (kv < KV_)   return Vp[(size_t)kv * HD_ + d];
    if (kv < KVTOT) return Vn[(kv - KV_) * HD_ + d];
    return 0.0f;
}

// ---------------------------------------------------------------------------
// Kernel 3: flash attention.  One block (4 waves) per (b,h).  Each wave takes
// KV chunks of 32 rows (strided by 4) with online softmax, then waves merge
// (m, l, O) partials through LDS.
//   S^T = K_chunk(16x128) x Q^T(128x16): lane = q column -> per-q softmax
//   needs only shfl_xor(16).  P^T staged via LDS; O^T += V^T x P^T (K=32).
// ---------------------------------------------------------------------------
__global__ void __launch_bounds__(128) attn_kernel(
    const float* __restrict__ qb, const float* __restrict__ kb,
    const float* __restrict__ vb,
    const float* __restrict__ pk, const float* __restrict__ pv,
    float* __restrict__ ctx)
{
    __shared__ float Plds[4][32][16];
    __shared__ float Mw[4][16];
    __shared__ float Lw[4][16];
    __shared__ float Ocomb[4][SS_][HD_];

    const int bh = blockIdx.x;                // b*NH + h
    const int bi = bh >> 4, h = bh & 15;
    const int tid = threadIdx.x;
    const int w = tid >> 5, lane = tid & 31;
    const int hf = lane >> 4, l15 = lane & 15;

    const float* Q  = qb + bh * SS_ * HD_;
    const float* Kn = kb + bh * SS_ * HD_;
    const float* Vn = vb + bh * SS_ * HD_;
    const float* Kp = pk + (size_t)bh * KV_ * HD_;
    const float* Vp = pv + (size_t)bh * KV_ * HD_;

    // Q^T B-fragments, one per 32-wide K step; q rows >= 8 zero-padded.
    v16bf qf[4];
#pragma unroll
    for (int ks = 0; ks < 4; ++ks) {
        if (l15 < SS_) {
            const float* qp = Q + l15 * HD_ + ks * 32 + hf * 16;
            const float4 t0 = *(const float4*)(qp);
            const float4 t1 = *(const float4*)(qp + 4);
            const float4 t2 = *(const float4*)(qp + 8);
            const float4 t3 = *(const float4*)(qp + 12);
            qf[ks][0]=(__bf16)t0.x;  qf[ks][1]=(__bf16)t0.y;
            qf[ks][2]=(__bf16)t0.z;  qf[ks][3]=(__bf16)t0.w;
            qf[ks][4]=(__bf16)t1.x;  qf[ks][5]=(__bf16)t1.y;
            qf[ks][6]=(__bf16)t1.z;  qf[ks][7]=(__bf16)t1.w;
            qf[ks][8]=(__bf16)t2.x;  qf[ks][9]=(__bf16)t2.y;
            qf[ks][10]=(__bf16)t2.z; qf[ks][11]=(__bf16)t2.w;
            qf[ks][12]=(__bf16)t3.x; qf[ks][13]=(__bf16)t3.y;
            qf[ks][14]=(__bf16)t3.z; qf[ks][15]=(__bf16)t3.w;
        } else {
#pragma unroll
            for (int j = 0; j < 16; ++j) qf[ks][j] = (__bf16)0.0f;
        }
    }

    float m_i = -INFINITY, l_i = 0.0f;
    v8f o[8];
#pragma unroll
    for (int n = 0; n < 8; ++n)
#pragma unroll
        for (int r = 0; r < 8; ++r) o[n][r] = 0.0f;

    const float scale = 0.08838834764831845f;   // 1/sqrt(128)
    const int nch = (KVTOT + 31) / 32;          // 129

    for (int c = w; c < nch; c += 4) {
        const int kvb = c * 32;
        float vals[16];
        float smax = -INFINITY;

#pragma unroll
        for (int sub = 0; sub < 2; ++sub) {
            v8f st;
#pragma unroll
            for (int r = 0; r < 8; ++r) st[r] = 0.0f;

            const int krow = kvb + sub * 16 + l15;
            const float* krp = nullptr;
            if (krow < KV_)        krp = Kp + (size_t)krow * HD_;
            else if (krow < KVTOT) krp = Kn + (krow - KV_) * HD_;

#pragma unroll
            for (int ks = 0; ks < 4; ++ks) {
                v16bf a;
                if (krp) {
                    const float* p0 = krp + ks * 32 + hf * 16;
                    const float4 s0 = *(const float4*)(p0);
                    const float4 s1 = *(const float4*)(p0 + 4);
                    const float4 s2 = *(const float4*)(p0 + 8);
                    const float4 s3 = *(const float4*)(p0 + 12);
                    a[0]=(__bf16)s0.x;  a[1]=(__bf16)s0.y;
                    a[2]=(__bf16)s0.z;  a[3]=(__bf16)s0.w;
                    a[4]=(__bf16)s1.x;  a[5]=(__bf16)s1.y;
                    a[6]=(__bf16)s1.z;  a[7]=(__bf16)s1.w;
                    a[8]=(__bf16)s2.x;  a[9]=(__bf16)s2.y;
                    a[10]=(__bf16)s2.z; a[11]=(__bf16)s2.w;
                    a[12]=(__bf16)s3.x; a[13]=(__bf16)s3.y;
                    a[14]=(__bf16)s3.z; a[15]=(__bf16)s3.w;
                } else {
#pragma unroll
                    for (int j = 0; j < 16; ++j) a[j] = (__bf16)0.0f;
                }
                st = __builtin_amdgcn_wmma_f32_16x16x32_bf16(false, a, false, qf[ks],
                                                             (short)0, st, false, false);
            }
            // D layout: lane owns q = l15; kv = sub*16 + r + 8*(lane>>4)
#pragma unroll
            for (int r = 0; r < 8; ++r) {
                const int kvloc = sub * 16 + r + 8 * hf;
                const float v = (kvb + kvloc < KVTOT) ? st[r] * scale : -INFINITY;
                vals[sub * 8 + r] = v;
                smax = fmaxf(smax, v);
            }
        }

        // per-q chunk max / sum: other 16 kv values live in lane ^ 16
        smax = fmaxf(smax, __shfl_xor(smax, 16, 32));
        const float mnew  = fmaxf(m_i, smax);
        const float alpha = __expf(m_i - mnew);
        float csum = 0.0f;
#pragma unroll
        for (int i2 = 0; i2 < 16; ++i2) {
            const float p = __expf(vals[i2] - mnew);
            vals[i2] = p;
            csum += p;
        }
        csum += __shfl_xor(csum, 16, 32);
        l_i = l_i * alpha + csum;
        m_i = mnew;
#pragma unroll
        for (int n = 0; n < 8; ++n) o[n] *= alpha;

        // stage P^T (32 kv x 16 q) in this wave's LDS tile
#pragma unroll
        for (int sub = 0; sub < 2; ++sub)
#pragma unroll
            for (int r = 0; r < 8; ++r)
                Plds[w][sub * 16 + r + 8 * hf][l15] = vals[sub * 8 + r];
        // intra-wave cross-lane LDS RAW: DS ops are in-order per wave,
        // but we must drain DScnt before re-reading in a new lane layout.
        asm volatile("s_wait_dscnt 0" ::: "memory");

        // B fragment: element j -> kv = 16*(lane>>4) + j, col q = l15
        v16bf pb;
#pragma unroll
        for (int j = 0; j < 16; ++j)
            pb[j] = (__bf16)Plds[w][hf * 16 + j][l15];

        // O^T tiles: A = V^T (16 d x 32 kv), per-lane row d = n*16 + l15
#pragma unroll
        for (int n = 0; n < 8; ++n) {
            const int d = n * 16 + l15;
            v16bf va;
#pragma unroll
            for (int j = 0; j < 8; ++j) {
                const int kv1 = kvb + hf * 16 + j;
                const int kv2 = kvb + hf * 16 + 8 + j;
                va[j]     = (__bf16)load_v_elem(Vp, Vn, kv1, d);
                va[j + 8] = (__bf16)load_v_elem(Vp, Vn, kv2, d);
            }
            o[n] = __builtin_amdgcn_wmma_f32_16x16x32_bf16(false, va, false, pb,
                                                           (short)0, o[n], false, false);
        }
    }

    // ---- merge the 4 waves' partials ----
    if (hf == 0) { Mw[w][l15] = m_i; Lw[w][l15] = l_i; }
    if (l15 < SS_) {
#pragma unroll
        for (int n = 0; n < 8; ++n)
#pragma unroll
            for (int r = 0; r < 8; ++r)
                Ocomb[w][l15][n * 16 + r + 8 * hf] = o[n][r];
    }
    __syncthreads();

    for (int t = tid; t < SS_ * HD_; t += 128) {
        const int q = t >> 7, d = t & 127;
        float M = Mw[0][q];
#pragma unroll
        for (int ww = 1; ww < 4; ++ww) M = fmaxf(M, Mw[ww][q]);
        float osum = 0.0f, ls = 0.0f;
#pragma unroll
        for (int ww = 0; ww < 4; ++ww) {
            const float e = __expf(Mw[ww][q] - M);
            osum += Ocomb[ww][q][d] * e;
            ls   += Lw[ww][q] * e;
        }
        ctx[(bi * SS_ + q) * (NH_ * HD_) + h * HD_ + d] = osum / ls;
    }
}

// ---------------------------------------------------------------------------
// Kernel 4: output projection ctx[128x2048] * Wo[2048x2048] -> out.
// ---------------------------------------------------------------------------
__global__ void __launch_bounds__(128) out_gemm_kernel(
    const float* __restrict__ ctx, const float* __restrict__ Wo,
    float* __restrict__ out)
{
    const int wave = threadIdx.x >> 5;
    const int tile = blockIdx.x * 4 + wave;   // 0..1023
    const int nt = tile >> 3;                 // 0..127
    const int mt = tile & 7;
    const int m0 = mt * 16, n0 = nt * 16;

    v8f acc = gemm_tile_bf16(ctx, Wo, m0, n0);

    const int lane = threadIdx.x & 31;
    const int hf = lane >> 4, l15 = lane & 15;
#pragma unroll
    for (int r = 0; r < 8; ++r) {
        const int m = m0 + r + 8 * hf;
        out[(size_t)m * HH_ + n0 + l15] = acc[r];
    }
}

// ---------------------------------------------------------------------------
extern "C" void kernel_launch(void* const* d_in, const int* in_sizes, int n_in,
                              void* d_out, int out_size, void* d_ws, size_t ws_size,
                              hipStream_t stream)
{
    const float* x  = (const float*)d_in[0];
    const float* Wq = (const float*)d_in[1];
    const float* Wk = (const float*)d_in[2];
    const float* Wv = (const float*)d_in[3];
    const float* Wo = (const float*)d_in[4];
    const float* pk = (const float*)d_in[5];
    const float* pv = (const float*)d_in[6];

    float* ws  = (float*)d_ws;
    float* qb  = ws;                    // [B,nh,S,hd]  1 MB
    float* kb  = ws + 262144;           // [B,nh,S,hd]  1 MB
    float* vb  = ws + 2 * 262144;       // [B,nh,S,hd]  1 MB
    float* ctx = ws + 3 * 262144;       // [B,S,nh*hd]  1 MB

    qkv_gemm_kernel<<<768, 128, 0, stream>>>(x, Wq, Wk, Wv, qb, kb, vb);
    rope_kernel<<<1024, 256, 0, stream>>>(qb, kb);
    attn_kernel<<<BB_ * NH_, 128, 0, stream>>>(qb, kb, vb, pk, pv, ctx);
    out_gemm_kernel<<<256, 128, 0, stream>>>(ctx, Wo, (float*)d_out);
}
